// UTUV1Attention_20641612824697
// MI455X (gfx1250) — compile-verified, matrix operand order checked by reading, not run on hardware
//
#include <hip/hip_runtime.h>
#include <stdint.h>

typedef __attribute__((ext_vector_type(16))) _Float16 v16h;
typedef __attribute__((ext_vector_type(8)))  _Float16 v8h;
typedef __attribute__((ext_vector_type(8)))  float    v8f;
typedef __attribute__((ext_vector_type(4)))  int      v4i;

#define S_LEN   2048
#define HID     2048
#define NHEAD   16
#define D_NOPE  128
#define D_ROPE  64
#define D_VDIM  128
#define D_QK    192
#define KV_R    512
#define SCALING 0.07216878364870323f   /* 192^-0.5 */
#define NEGBIG  (-1.0e9f)

// ---------------------------------------------------------------------------
// Async global->LDS copy (16B per lane), ASYNCcnt-tracked. The gfx1250
// builtin takes typed int4 pointers in AS1 (global) / AS3 (LDS).
// ---------------------------------------------------------------------------
__device__ __forceinline__ void async_copy_b128(const _Float16* g, _Float16* l) {
#if __has_builtin(__builtin_amdgcn_global_load_async_to_lds_b128)
  __builtin_amdgcn_global_load_async_to_lds_b128(
      (__attribute__((address_space(1))) v4i*)(void*)g,
      (__attribute__((address_space(3))) v4i*)(void*)l, 0, 0);
#else
  *(v8h*)l = *(const v8h*)g;
#endif
}

template <int N>
__device__ __forceinline__ void wait_async() {
#if __has_builtin(__builtin_amdgcn_s_wait_asynccnt)
  __builtin_amdgcn_s_wait_asynccnt(N);
#else
  asm volatile("s_wait_asynccnt %0" ::"i"(N) : "memory");
#endif
}

// ---------------------------------------------------------------------------
// Fragment loader: per-lane K-packed layout for 16-bit WMMA A/B operands
// (CDNA5 ISA 7.12.2). Row-major [rows][ld] with K contiguous -> two 16-byte
// chunks per lane. Works for global or LDS base pointers (inlined).
// ---------------------------------------------------------------------------
__device__ __forceinline__ v16h load_frag(const _Float16* base, int ld,
                                          int row, int k0, int lane) {
  const _Float16* p = base + (size_t)row * ld + k0 + ((lane & 16) ? 8 : 0);
  v8h lo = *(const v8h*)(p);
  v8h hi = *(const v8h*)(p + 16);
  v16h r;
#pragma unroll
  for (int i = 0; i < 8; ++i) { r[i] = lo[i]; r[i + 8] = hi[i]; }
  return r;
}

// ---------------------------------------------------------------------------
// f32 -> f16 conversion (grid-stride)
// ---------------------------------------------------------------------------
__global__ void cvt_f32_f16(const float* __restrict__ in,
                            _Float16* __restrict__ out, int n) {
  int i = blockIdx.x * blockDim.x + threadIdx.x;
  int stride = gridDim.x * blockDim.x;
  for (; i < n; i += stride) out[i] = (_Float16)in[i];
}

// ---------------------------------------------------------------------------
// WMMA GEMM: C[M,N](f32) = A[M,K](f16 row-major) @ W[N,K]^T (f16).
// Block = 8 waves = 128x64 C tile. The 64x32 B strip for each K-chunk is
// staged once per block into LDS by async copies (double-buffered, one 16B
// chunk per thread) and shared by all 8 waves via ds_load_b128. A-fragments
// come straight from global. M%128==0, N%64==0, K%32==0.
// ---------------------------------------------------------------------------
__global__ __launch_bounds__(256)
void gemm_wmma(const _Float16* __restrict__ A, const _Float16* __restrict__ W,
               float* __restrict__ C, int M, int N, int K) {
  __shared__ __align__(16) _Float16 bsh[2][64 * 32];

  const int t    = threadIdx.x;
  const int lane = t & 31;
  const int wave = t >> 5;
  const int col  = lane & 15;
  const int half = lane >> 4;

  const int nstrips = N >> 6;
  const int n0 = (blockIdx.x % nstrips) << 6;
  const int m0 = (blockIdx.x / nstrips) * 128 + wave * 16;

  // this thread's 16B slice of the 64x32 B strip
  const int nloc  = t >> 2;           // 0..63
  const int piece = (t & 3) * 8;      // f16 offset within row
  const _Float16* wsrc = W + (size_t)(n0 + nloc) * K + piece;
  _Float16* ldst[2] = { &bsh[0][nloc * 32 + piece], &bsh[1][nloc * 32 + piece] };

  v8f acc[4];
#pragma unroll
  for (int j = 0; j < 4; ++j)
#pragma unroll
    for (int v = 0; v < 8; ++v) acc[j][v] = 0.0f;

  async_copy_b128(wsrc, ldst[0]);     // prologue: stage k-chunk 0
  int buf = 0;
  for (int k0 = 0; k0 < K; k0 += 32, buf ^= 1) {
    const bool more = (k0 + 32) < K;
    if (more) {                        // prefetch next chunk into other buffer
      async_copy_b128(wsrc + (k0 + 32), ldst[buf ^ 1]);
      wait_async<1>();                 // current chunk done, prefetch in flight
    } else {
      wait_async<0>();
    }
    __syncthreads();

    v16h a = load_frag(A, K, m0 + col, k0, lane);
    const _Float16* bb = bsh[buf];
#pragma unroll
    for (int j = 0; j < 4; ++j) {
      v16h b = load_frag(bb, 32, j * 16 + col, 0, lane);
      acc[j] = __builtin_amdgcn_wmma_f32_16x16x32_f16(
          false, a, false, b, (short)0, acc[j], false, false);
    }
    __syncthreads();                   // done reading before same buf refilled
  }

#pragma unroll
  for (int j = 0; j < 4; ++j)
#pragma unroll
    for (int v = 0; v < 8; ++v)
      C[(size_t)(m0 + v + 8 * half) * N + n0 + j * 16 + col] = acc[j][v];
}

// ---------------------------------------------------------------------------
// RMSNorm over ckv[:,0:512] -> cnorm (f16) + RoPE on ckv[:,512:576] -> krot.
// ---------------------------------------------------------------------------
__global__ __launch_bounds__(256)
void rms_rope_k(const float* __restrict__ ckv, const float* __restrict__ lnw,
                const float* __restrict__ cosp, const float* __restrict__ sinp,
                _Float16* __restrict__ cnorm, float* __restrict__ krot) {
  const int s = blockIdx.x;
  const int t = threadIdx.x;
  const float* row = ckv + (size_t)s * (KV_R + D_ROPE);
  __shared__ float red[256];

  float x0 = row[t];
  float x1 = row[t + 256];
  red[t] = x0 * x0 + x1 * x1;
  __syncthreads();
  for (int k = 128; k > 0; k >>= 1) {
    if (t < k) red[t] += red[t + k];
    __syncthreads();
  }
  float inv = rsqrtf(red[0] * (1.0f / (float)KV_R) + 1e-6f);
  cnorm[(size_t)s * KV_R + t]       = (_Float16)(x0 * inv * lnw[t]);
  cnorm[(size_t)s * KV_R + 256 + t] = (_Float16)(x1 * inv * lnw[t + 256]);

  if (t < 32) {
    float a = row[KV_R + 2 * t];
    float b = row[KV_R + 2 * t + 1];
    float c0 = cosp[s * D_ROPE + t],      s0 = sinp[s * D_ROPE + t];
    float c1 = cosp[s * D_ROPE + 32 + t], s1 = sinp[s * D_ROPE + 32 + t];
    krot[s * D_ROPE + t]      = a * c0 - b * s0;
    krot[s * D_ROPE + 32 + t] = b * c1 + a * s1;
  }
}

// ---------------------------------------------------------------------------
// RoPE(q) + head-major repack + fold in 1/sqrt(d_qk).
// ---------------------------------------------------------------------------
__global__ __launch_bounds__(64)
void rope_q(const float* __restrict__ Q, const float* __restrict__ cosp,
            const float* __restrict__ sinp, _Float16* __restrict__ Qh) {
  const int h = blockIdx.x & (NHEAD - 1);
  const int s = blockIdx.x >> 4;
  const int t = threadIdx.x;
  const float* qrow = Q + (size_t)s * (NHEAD * D_QK) + h * D_QK;
  _Float16* out = Qh + ((size_t)h * S_LEN + s) * D_QK;

  out[t]      = (_Float16)(qrow[t] * SCALING);
  out[t + 64] = (_Float16)(qrow[t + 64] * SCALING);
  if (t < 32) {
    float a = qrow[D_NOPE + 2 * t];
    float b = qrow[D_NOPE + 2 * t + 1];
    float c0 = cosp[s * D_ROPE + t],      s0 = sinp[s * D_ROPE + t];
    float c1 = cosp[s * D_ROPE + 32 + t], s1 = sinp[s * D_ROPE + 32 + t];
    out[D_NOPE + t]      = (_Float16)((a * c0 - b * s0) * SCALING);
    out[D_NOPE + 32 + t] = (_Float16)((b * c1 + a * s1) * SCALING);
  }
}

// ---------------------------------------------------------------------------
// kv f32 [S, NH*256] -> K f16 [NH][S][192] and V^T f16 [NH][128][S].
// ---------------------------------------------------------------------------
__global__ __launch_bounds__(128)
void rearrange_kv(const float* __restrict__ kv, const float* __restrict__ krot,
                  _Float16* __restrict__ Kh, _Float16* __restrict__ Vt) {
  const int h = blockIdx.x & (NHEAD - 1);
  const int s = blockIdx.x >> 4;
  const int d = threadIdx.x;
  const float* row = kv + (size_t)s * (NHEAD * (D_NOPE + D_VDIM)) + h * (D_NOPE + D_VDIM);
  Kh[((size_t)h * S_LEN + s) * D_QK + d] = (_Float16)row[d];
  Vt[((size_t)h * D_VDIM + d) * S_LEN + s] = (_Float16)row[D_NOPE + d];
  if (d < D_ROPE)
    Kh[((size_t)h * S_LEN + s) * D_QK + D_NOPE + d] = (_Float16)krot[s * D_ROPE + d];
}

// ---------------------------------------------------------------------------
// Flash attention. Block = one head x 128-row query supertile (8 waves, one
// 16-row q-tile each). K (32x192) and V^T (128x32) tiles for each 32-key
// block are staged into double-buffered LDS by async copies and shared by all
// waves. Trip count is block-uniform (waves past their causal frontier just
// compute fully-masked scores -> p=0), so barriers are safe and EXEC is
// all-ones at every WMMA. P (16x32) is staged through LDS to convert the f32
// C-layout into the f16 A-layout (DS ops are in-order per wave).
// ---------------------------------------------------------------------------
__global__ __launch_bounds__(256)
void flash_attn(const _Float16* __restrict__ Qh, const _Float16* __restrict__ Kh,
                const _Float16* __restrict__ Vt, _Float16* __restrict__ attn) {
  __shared__ __align__(16) _Float16 ksh[2][32 * D_QK];    // 24 KB
  __shared__ __align__(16) _Float16 vsh[2][D_VDIM * 32];  // 16 KB
  __shared__ __align__(16) _Float16 pbuf[8][16 * 32];     //  8 KB

  const int t    = threadIdx.x;
  const int lane = t & 31;
  const int wave = t >> 5;
  const int col  = lane & 15;
  const int half = lane >> 4;

  const int h   = blockIdx.x >> 4;         // 16 supertiles per head
  const int q0b = (blockIdx.x & 15) << 7;  // block query base
  const int q0  = q0b + wave * 16;         // this wave's q-tile

  const _Float16* Kg = Kh + (size_t)h * S_LEN * D_QK;
  const _Float16* Vg = Vt + (size_t)h * D_VDIM * S_LEN;

  // preload Q fragments (16 x 192)
  const _Float16* Qb = Qh + ((size_t)h * S_LEN + q0) * D_QK;
  v16h qf[6];
#pragma unroll
  for (int u = 0; u < 6; ++u) qf[u] = load_frag(Qb, D_QK, col, u * 32, lane);

  v8f o[8];
  float m_i[8], l_i[8];
#pragma unroll
  for (int j = 0; j < 8; ++j)
#pragma unroll
    for (int v = 0; v < 8; ++v) o[j][v] = 0.0f;
#pragma unroll
  for (int v = 0; v < 8; ++v) { m_i[v] = -3.0e38f; l_i[v] = 0.0f; }

  _Float16* pb = pbuf[wave];
  const int nkb = (q0b + 128) / 32;        // key blocks for this supertile

  // stage key block `kbase` into buffer b: 5 async b128 chunks per thread
  auto stage = [&](int kbase, int b) {
#pragma unroll
    for (int i = 0; i < 3; ++i) {          // K tile: 32 rows x 384B = 768 chunks
      int c = t + i * 256;
      int row = c / 24, piece = (c % 24) * 8;
      async_copy_b128(Kg + (size_t)(kbase + row) * D_QK + piece,
                      &ksh[b][row * D_QK + piece]);
    }
#pragma unroll
    for (int i = 0; i < 2; ++i) {          // V tile: 128 rows x 64B = 512 chunks
      int c = t + i * 256;
      int d = c >> 2, piece = (c & 3) * 8;
      async_copy_b128(Vg + (size_t)d * S_LEN + kbase + piece,
                      &vsh[b][d * 32 + piece]);
    }
  };

  stage(0, 0);
  int buf = 0;
  for (int kb = 0; kb < nkb; ++kb, buf ^= 1) {
    const int kbase = kb * 32;
    const bool more = (kb + 1) < nkb;
    if (more) {
      stage(kbase + 32, buf ^ 1);
      wait_async<5>();                     // current tile done, prefetch pending
    } else {
      wait_async<0>();
    }
    __syncthreads();

    const _Float16* kt = ksh[buf];
    const _Float16* vt = vsh[buf];

    v8f s0, s1;
#pragma unroll
    for (int v = 0; v < 8; ++v) { s0[v] = 0.0f; s1[v] = 0.0f; }
#pragma unroll
    for (int u = 0; u < 6; ++u) {
      v16h b0 = load_frag(kt, D_QK, col, u * 32, lane);
      v16h b1 = load_frag(kt + 16 * D_QK, D_QK, col, u * 32, lane);
      s0 = __builtin_amdgcn_wmma_f32_16x16x32_f16(false, qf[u], false, b0,
                                                  (short)0, s0, false, false);
      s1 = __builtin_amdgcn_wmma_f32_16x16x32_f16(false, qf[u], false, b1,
                                                  (short)0, s1, false, false);
    }
    // causal mask (per-lane arithmetic; rows fully behind the frontier get
    // all-NEG rows -> p == 0, stats unchanged)
    const int qrow0 = q0 + 8 * half;
#pragma unroll
    for (int v = 0; v < 8; ++v) {
      if (kbase + col      > qrow0 + v) s0[v] = NEGBIG;
      if (kbase + 16 + col > qrow0 + v) s1[v] = NEGBIG;
    }
    // online softmax
    float mnew[8];
#pragma unroll
    for (int v = 0; v < 8; ++v) mnew[v] = fmaxf(m_i[v], fmaxf(s0[v], s1[v]));
#pragma unroll
    for (int mk = 8; mk >= 1; mk >>= 1)
#pragma unroll
      for (int v = 0; v < 8; ++v)
        mnew[v] = fmaxf(mnew[v], __shfl_xor(mnew[v], mk, 32));

    float p0[8], p1[8], rs[8], scl[8];
#pragma unroll
    for (int v = 0; v < 8; ++v) {
      scl[v] = __expf(m_i[v] - mnew[v]);
      p0[v] = __expf(s0[v] - mnew[v]);
      p1[v] = __expf(s1[v] - mnew[v]);
      rs[v] = p0[v] + p1[v];
    }
#pragma unroll
    for (int mk = 8; mk >= 1; mk >>= 1)
#pragma unroll
      for (int v = 0; v < 8; ++v) rs[v] += __shfl_xor(rs[v], mk, 32);
#pragma unroll
    for (int v = 0; v < 8; ++v) {
      l_i[v] = l_i[v] * scl[v] + rs[v];
      m_i[v] = mnew[v];
    }
#pragma unroll
    for (int j = 0; j < 8; ++j)
#pragma unroll
      for (int v = 0; v < 8; ++v) o[j][v] *= scl[v];

    // stage P through LDS: C-layout f32 -> A-layout f16
#pragma unroll
    for (int v = 0; v < 8; ++v) {
      pb[(8 * half + v) * 32 + col]      = (_Float16)p0[v];
      pb[(8 * half + v) * 32 + 16 + col] = (_Float16)p1[v];
    }
    asm volatile("s_wait_dscnt 0x0" ::: "memory");
    v16h pf = load_frag(pb, 32, col, 0, lane);

    // P @ V from LDS V^T tile
#pragma unroll
    for (int j = 0; j < 8; ++j) {
      v16h bv = load_frag(vt, 32, j * 16 + col, 0, lane);
      o[j] = __builtin_amdgcn_wmma_f32_16x16x32_f16(false, pf, false, bv,
                                                    (short)0, o[j], false, false);
    }
    __syncthreads();                       // all waves done with buf
  }

  // epilogue: normalize and scatter to [s][h*128 + d]
#pragma unroll
  for (int j = 0; j < 8; ++j)
#pragma unroll
    for (int v = 0; v < 8; ++v) {
      float val = o[j][v] / l_i[v];
      attn[(size_t)(q0 + v + 8 * half) * (NHEAD * D_VDIM) + h * D_VDIM +
           j * 16 + col] = (_Float16)val;
    }
}

// ---------------------------------------------------------------------------
// host-side orchestration
// ---------------------------------------------------------------------------
extern "C" void kernel_launch(void* const* d_in, const int* in_sizes, int n_in,
                              void* d_out, int out_size, void* d_ws, size_t ws_size,
                              hipStream_t stream) {
  (void)in_sizes; (void)n_in; (void)out_size; (void)ws_size;
  const float* hs   = (const float*)d_in[0];
  const float* cosp = (const float*)d_in[1];
  const float* sinp = (const float*)d_in[2];
  /* d_in[3] = attention_mask: exactly causal -> handled analytically */
  const float* qw   = (const float*)d_in[4];
  const float* kvaw = (const float*)d_in[5];
  const float* lnw  = (const float*)d_in[6];
  const float* kvbw = (const float*)d_in[7];
  const float* ow   = (const float*)d_in[8];
  float* out = (float*)d_out;

  char* ws = (char*)d_ws;
  size_t off = 0;
  auto alloc = [&](size_t bytes) -> void* {
    void* p = (void*)(ws + off);
    off += (bytes + 255) & ~(size_t)255;
    return p;
  };

  _Float16* hs16    = (_Float16*)alloc((size_t)S_LEN * HID * 2);
  _Float16* qw16    = (_Float16*)alloc((size_t)NHEAD * D_QK * HID * 2);
  _Float16* kvaw16  = (_Float16*)alloc((size_t)(KV_R + D_ROPE) * HID * 2);
  _Float16* kvbw16  = (_Float16*)alloc((size_t)NHEAD * (D_NOPE + D_VDIM) * KV_R * 2);
  _Float16* ow16    = (_Float16*)alloc((size_t)HID * NHEAD * D_VDIM * 2);
  float*    Qf32    = (float*)alloc((size_t)S_LEN * NHEAD * D_QK * 4);
  float*    ckv     = (float*)alloc((size_t)S_LEN * (KV_R + D_ROPE) * 4);
  _Float16* cnorm16 = (_Float16*)alloc((size_t)S_LEN * KV_R * 2);
  float*    krot    = (float*)alloc((size_t)S_LEN * D_ROPE * 4);
  float*    kvf32   = (float*)alloc((size_t)S_LEN * NHEAD * (D_NOPE + D_VDIM) * 4);
  _Float16* Qh16    = (_Float16*)alloc((size_t)NHEAD * S_LEN * D_QK * 2);
  _Float16* Kh16    = (_Float16*)alloc((size_t)NHEAD * S_LEN * D_QK * 2);
  _Float16* Vt16    = (_Float16*)alloc((size_t)NHEAD * D_VDIM * S_LEN * 2);
  _Float16* attn16  = (_Float16*)alloc((size_t)S_LEN * NHEAD * D_VDIM * 2);

  // 1) f32 -> f16 conversions
  cvt_f32_f16<<<2048, 256, 0, stream>>>(hs, hs16, S_LEN * HID);
  cvt_f32_f16<<<2048, 256, 0, stream>>>(qw, qw16, NHEAD * D_QK * HID);
  cvt_f32_f16<<<2048, 256, 0, stream>>>(kvaw, kvaw16, (KV_R + D_ROPE) * HID);
  cvt_f32_f16<<<2048, 256, 0, stream>>>(kvbw, kvbw16, NHEAD * (D_NOPE + D_VDIM) * KV_R);
  cvt_f32_f16<<<2048, 256, 0, stream>>>(ow, ow16, HID * NHEAD * D_VDIM);

  // 2) Q = hs @ Wq^T   [2048 x 3072], K=2048
  {
    int M = S_LEN, N = NHEAD * D_QK, K = HID;
    gemm_wmma<<<(M / 128) * (N / 64), 256, 0, stream>>>(hs16, qw16, Qf32, M, N, K);
  }
  // 3) ckv = hs @ Wkva^T   [2048 x 576], K=2048
  {
    int M = S_LEN, N = KV_R + D_ROPE, K = HID;
    gemm_wmma<<<(M / 128) * (N / 64), 256, 0, stream>>>(hs16, kvaw16, ckv, M, N, K);
  }
  // 4) RMSNorm + RoPE(k)
  rms_rope_k<<<S_LEN, 256, 0, stream>>>(ckv, lnw, cosp, sinp, cnorm16, krot);
  // 5) kv = cnorm @ Wkvb^T   [2048 x 4096], K=512
  {
    int M = S_LEN, N = NHEAD * (D_NOPE + D_VDIM), K = KV_R;
    gemm_wmma<<<(M / 128) * (N / 64), 256, 0, stream>>>(cnorm16, kvbw16, kvf32, M, N, K);
  }
  // 6) RoPE(q) + repack + scaling fold
  rope_q<<<NHEAD * S_LEN, 64, 0, stream>>>(Qf32, cosp, sinp, Qh16);
  // 7) K/V rearrange (+V transpose)
  rearrange_kv<<<NHEAD * S_LEN, 128, 0, stream>>>(kvf32, krot, Kh16, Vt16);
  // 8) flash attention: 16 heads x 16 supertiles = 256 blocks
  flash_attn<<<NHEAD * (S_LEN / 128), 256, 0, stream>>>(Qh16, Kh16, Vt16, attn16);
  // 9) out = attn @ Wo^T   [2048 x 2048], K=2048
  {
    int M = S_LEN, N = HID, K = NHEAD * D_VDIM;
    gemm_wmma<<<(M / 128) * (N / 64), 256, 0, stream>>>(attn16, ow16, out, M, N, K);
  }
}
